// AttentionPooling_58780922413438
// MI455X (gfx1250) — matrix operand order repaired and checked by reading, not compiled, not run on hardware
//
#include <hip/hip_runtime.h>
#include <math.h>

// ---------------------------------------------------------------------------
// AttentionPooling for MI455X (gfx1250, wave32, WMMA + async global->LDS).
//
// Key algebra:
//   * V-projection folded through attention:  out_h = Wv_h * (sum_s a_s x_s) + bv_h
//   * RoPE folded into q:  q . rope(k) = qrot(s) . (Wk x + bk)
// so the only large GEMM is the K projection, done in bf16 WMMA
// (v_wmma_f32_16x16x32_bf16). Bulk staging copies use
// global_load_async_to_lds_b128 (ASYNCcnt) to bypass VGPRs.
// ---------------------------------------------------------------------------

#define DIM   768
#define NH    12
#define HDIM  64
#define HALF  32
#define LATD  256
#define NB    16
#define SEQ   8192
#define TS    32                      // positions per score-tile workgroup
#define LOG2_ROPE 13.287712379549449f // log2(10000)

typedef __attribute__((ext_vector_type(16))) __bf16 bf16x16;
typedef __attribute__((ext_vector_type(8)))  float  f32x8;

// LDS byte offset of a __shared__ object: generic shared-aperture addresses
// carry the LDS offset in their low 32 bits (ISA 10.2 aperture mapping).
__device__ __forceinline__ unsigned __lds_off(const void* p) {
    return (unsigned)(unsigned long long)p;
}

// Async DMA: 16 bytes global -> LDS per active lane (GLOBAL_LOAD_ASYNC_TO_LDS_B128,
// GV addressing: VDST = LDS byte offset, VADDR = 64-bit global address).
__device__ __forceinline__ void async_g2l_b128(unsigned lds_byte_off, const void* gaddr) {
    asm volatile("global_load_async_to_lds_b128 %0, %1, off"
                 :: "v"(lds_byte_off), "v"(gaddr)
                 : "memory");
}
__device__ __forceinline__ void wait_async0() {
    asm volatile("s_wait_asynccnt 0x0" ::: "memory");
}

// --------------------------- kernel 1: Wk -> bf16 ---------------------------
__global__ void k_cvt_wk(const float* __restrict__ Wk, __bf16* __restrict__ wkbf) {
    int i = blockIdx.x * 256 + threadIdx.x;
    if (i < DIM * DIM) wkbf[i] = (__bf16)Wk[i];
}

// --------------------------- kernel 2: q = Wq pool + bq ---------------------
__global__ void k_q(const float* __restrict__ pool, const float* __restrict__ Wq,
                    const float* __restrict__ bq, float* __restrict__ qv) {
    int n = blockIdx.x * 256 + threadIdx.x;
    if (n < DIM) {
        const float* wr = Wq + (size_t)n * DIM;
        float s = bq[n];
        for (int d = 0; d < DIM; ++d) s = fmaf(wr[d], pool[d], s);
        qv[n] = s; // RoPE at pos 0 is identity
    }
}

// ---------------- kernel 3: K-projection (WMMA) + rope-folded scores --------
__global__ __launch_bounds__(256, 1)
void k_scores(const float* __restrict__ x, const float* __restrict__ bk,
              const __bf16* __restrict__ wkbf, const float* __restrict__ qv,
              float* __restrict__ scores)
{
    __shared__ __align__(16) float Asf[TS * DIM];     // 98304 B: x tile (fp32, async DMA)
    __shared__ float qlds[DIM];                       // 3072 B
    __shared__ float csT[TS * HALF * 2];              // 8192 B: cos/sin per (s_loc, freq)
    __shared__ float acc[TS * NH];                    // 1536 B: per (s,h) score accum

    const int tid = threadIdx.x;
    const int b   = blockIdx.y;
    const int s0  = blockIdx.x * TS;

    // Kick off async DMA of the whole x tile (contiguous 96KB block) into LDS.
    {
        const char*    gsrc = (const char*)(x + (size_t)(b * SEQ + s0) * DIM);
        const unsigned lbas = __lds_off(Asf);
        for (int off = tid * 16; off < TS * DIM * 4; off += 256 * 16)
            async_g2l_b128(lbas + off, gsrc + off);
    }

    // Overlap: q vector, sincos table, accumulator init while DMA is in flight.
    for (int i = tid; i < DIM; i += 256) qlds[i] = qv[i];
    for (int i = tid; i < TS * HALF; i += 256) {
        int sl = i >> 5, jf = i & 31;
        float invf = exp2f(-(float)jf * (LOG2_ROPE / (float)HALF));
        float ang  = (float)(s0 + sl) * invf;
        csT[2 * i]     = cosf(ang);
        csT[2 * i + 1] = sinf(ang);
    }
    for (int i = tid; i < TS * NH; i += 256) acc[i] = 0.f;
    wait_async0();
    __syncthreads();

    const int wave = tid >> 5;
    const int lane = tid & 31;
    const int hi   = lane >> 4;   // 0: lanes 0-15, 1: lanes 16-31
    const int l15  = lane & 15;
    const int n0w  = wave * 96;   // 6 N-tiles of 16 per wave (8 waves cover 768)

    f32x8 c[2][6];
#pragma unroll
    for (int m = 0; m < 2; ++m)
#pragma unroll
        for (int t = 0; t < 6; ++t) c[m][t] = 0.f;

    // K[s,n] = sum_d x[s,d] * Wk[n,d];  A: 16(M=s) x 32(K=d), B: 32(K=d) x 16(N=n)
    for (int kk = 0; kk < 24; ++kk) {
        const int kb = kk * 32;
        bf16x16 a[2];
#pragma unroll
        for (int m = 0; m < 2; ++m) {
            // A k order used here (same as B): lanes 0-15 hold K = kb+0..15 of row l15,
            // lanes 16-31 hold K = kb+16..31. Convert fp32->bf16 inline
            // (v_cvt co-executes under the WMMA shadow).
            const float* Ar = Asf + (m * 16 + l15) * DIM + kb + hi * 16;
#pragma unroll
            for (int i = 0; i < 8; ++i) a[m][i]     = (__bf16)Ar[i];
#pragma unroll
            for (int i = 0; i < 8; ++i) a[m][8 + i] = (__bf16)Ar[8 + i];
        }
#pragma unroll
        for (int t = 0; t < 6; ++t) {
            // B k order used here (same as A): lanes 0-15 hold K = kb+0..15, lanes 16-31 K = kb+16..31
            // of column n; B[k][n] = Wk[n][k] is contiguous in the bf16 copy.
            const int n = n0w + t * 16 + l15;
            const bf16x16 bfrag = *(const bf16x16*)(wkbf + (size_t)n * DIM + kb + hi * 16);
#pragma unroll
            for (int m = 0; m < 2; ++m)
                c[m][t] = __builtin_amdgcn_wmma_f32_16x16x32_bf16(
                    false, a[m], false, bfrag, (short)0, c[m][t], false, false);
        }
    }

    // Contract K tile with rope-rotated q:  score(s,h) += qrot(s,n) * K(s,n).
    // C layout: lane l15 = column n; VGPR v = row v + 8*hi of the 16-tile.
#pragma unroll
    for (int t = 0; t < 6; ++t) {
        const int n  = n0w + t * 16 + l15;
        const int h  = n >> 6;
        const int j  = n & 63;
        const int jf = j & 31;
        const float qa = qlds[h * 64 + jf];
        const float qb = qlds[h * 64 + 32 + jf];
#pragma unroll
        for (int m = 0; m < 2; ++m) {
#pragma unroll
            for (int v = 0; v < 8; ++v) {
                const int   sl = m * 16 + hi * 8 + v;
                const float co = csT[2 * (sl * HALF + jf)];
                const float si = csT[2 * (sl * HALF + jf) + 1];
                const float qr = (j < HALF) ? fmaf(qa, co, qb * si)
                                            : fmaf(qb, co, -qa * si);
                atomicAdd(&acc[sl * NH + h], qr * c[m][t][v]);
            }
        }
    }
    __syncthreads();

    // Finalize: + qrot.bk, * 1/sqrt(64), + ALiBi bias.
    for (int e = tid; e < TS * NH; e += 256) {
        const int sl = e / NH;
        const int h  = e - sl * NH;
        float bkdot = 0.f;
        for (int jf = 0; jf < HALF; ++jf) {
            const float co = csT[2 * (sl * HALF + jf)];
            const float si = csT[2 * (sl * HALF + jf) + 1];
            const float qa = qlds[h * 64 + jf];
            const float qb = qlds[h * 64 + 32 + jf];
            bkdot = fmaf(fmaf(qa, co,  qb * si), bk[h * 64 + jf],      bkdot);
            bkdot = fmaf(fmaf(qb, co, -qa * si), bk[h * 64 + 32 + jf], bkdot);
        }
        const float slope = exp2f(-2.f * (float)(h + 1) / 3.f);
        const float sg    = (float)(s0 + sl);
        scores[((size_t)(b * NH + h)) * SEQ + s0 + sl] =
            0.125f * (acc[e] + bkdot) - slope * sg;
    }
}

// --------------------------- kernel 4: softmax per (b,h) --------------------
__global__ __launch_bounds__(256)
void k_softmax(float* __restrict__ scores) {
    float* row = scores + (size_t)blockIdx.x * SEQ;
    __shared__ float red[256];
    const int tid = threadIdx.x;
    float m = -1e30f;
    for (int i = tid; i < SEQ; i += 256) m = fmaxf(m, row[i]);
    red[tid] = m; __syncthreads();
    for (int off = 128; off > 0; off >>= 1) {
        if (tid < off) red[tid] = fmaxf(red[tid], red[tid + off]);
        __syncthreads();
    }
    m = red[0]; __syncthreads();
    float s = 0.f;
    for (int i = tid; i < SEQ; i += 256) {
        float e = __expf(row[i] - m);
        row[i] = e; s += e;
    }
    red[tid] = s; __syncthreads();
    for (int off = 128; off > 0; off >>= 1) {
        if (tid < off) red[tid] += red[tid + off];
        __syncthreads();
    }
    const float inv = 1.f / red[0];
    for (int i = tid; i < SEQ; i += 256) row[i] *= inv;
}

// --------------------------- kernel 5: zero pooled_x ------------------------
__global__ void k_zero(float* __restrict__ p, int n) {
    int i = blockIdx.x * 256 + threadIdx.x;
    if (i < n) p[i] = 0.f;
}

// ------------------- kernel 6: pooled_x = sum_s attn * x --------------------
__global__ __launch_bounds__(256)
void k_pool(const float* __restrict__ x, const float* __restrict__ attn,
            float* __restrict__ pooledx) {
    __shared__ __align__(16) float at[NH * 512];      // 24 KB attn chunk (async DMA)
    const int tid = threadIdx.x;
    const int b   = blockIdx.y;
    const int s0  = blockIdx.x * 512;

    {
        const unsigned lbas = __lds_off(at);
        for (int i = tid * 4; i < NH * 512; i += 256 * 4) {
            int h = i >> 9, si = i & 511;
            async_g2l_b128(lbas + i * 4,
                           attn + ((size_t)(b * NH + h)) * SEQ + s0 + si);
        }
    }
    wait_async0();
    __syncthreads();

    float accr[NH][3];
#pragma unroll
    for (int h = 0; h < NH; ++h)
#pragma unroll
        for (int r = 0; r < 3; ++r) accr[h][r] = 0.f;

    for (int si = 0; si < 512; ++si) {
        const float* xr = x + ((size_t)(b * SEQ + s0 + si)) * DIM;
        if (si < 511) __builtin_prefetch(xr + DIM + tid * 3, 0, 1);
#pragma unroll
        for (int r = 0; r < 3; ++r) {
            const float xv = xr[tid + r * 256];
#pragma unroll
            for (int h = 0; h < NH; ++h)
                accr[h][r] = fmaf(at[h * 512 + si], xv, accr[h][r]);
        }
    }
    float* dst = pooledx + (size_t)b * NH * DIM;
#pragma unroll
    for (int h = 0; h < NH; ++h)
#pragma unroll
        for (int r = 0; r < 3; ++r)
            atomicAdd(&dst[h * DIM + tid + r * 256], accr[h][r]);
}

// ------------ kernel 7: Wv fold + Wo + Wl head (small matvecs) --------------
__global__ __launch_bounds__(256)
void k_head(const float* __restrict__ pooledx,
            const float* __restrict__ Wv, const float* __restrict__ bv,
            const float* __restrict__ Wo, const float* __restrict__ bo,
            const float* __restrict__ Wl, const float* __restrict__ bl,
            float* __restrict__ out) {
    __shared__ __align__(16) float px[NH * DIM];      // 36 KB (async DMA)
    __shared__ float ov[DIM];
    __shared__ float pv[DIM];
    const int tid = threadIdx.x;
    const int b   = blockIdx.x;

    {
        const unsigned lbas = __lds_off(px);
        const float*   gsrc = pooledx + (size_t)b * NH * DIM;
        for (int i = tid * 4; i < NH * DIM; i += 256 * 4)
            async_g2l_b128(lbas + i * 4, gsrc + i);
    }
    wait_async0();
    __syncthreads();

    for (int m = tid; m < DIM; m += 256) {
        const float* wr = Wv + (size_t)m * DIM;
        const float* pr = px + (m >> 6) * DIM;
        float s = bv[m];
        for (int d = 0; d < DIM; ++d) s = fmaf(wr[d], pr[d], s);
        ov[m] = s;
    }
    __syncthreads();
    for (int i = tid; i < DIM; i += 256) {
        const float* wr = Wo + (size_t)i * DIM;
        float s = bo[i];
        for (int m = 0; m < DIM; ++m) s = fmaf(wr[m], ov[m], s);
        pv[i] = s;
    }
    __syncthreads();
    for (int o = tid; o < LATD; o += 256) {
        const float* wr = Wl + (size_t)o * DIM;
        float s = bl[o];
        for (int i = 0; i < DIM; ++i) s = fmaf(wr[i], pv[i], s);
        out[(size_t)b * LATD + o] = s;
    }
}

// ---------------------------------------------------------------------------
extern "C" void kernel_launch(void* const* d_in, const int* in_sizes, int n_in,
                              void* d_out, int out_size, void* d_ws, size_t ws_size,
                              hipStream_t stream) {
    (void)in_sizes; (void)n_in; (void)out_size; (void)ws_size;
    const float* x    = (const float*)d_in[0];
    const float* pool = (const float*)d_in[1];
    const float* Wq   = (const float*)d_in[2];
    const float* bq   = (const float*)d_in[3];
    const float* Wk   = (const float*)d_in[4];
    const float* bk   = (const float*)d_in[5];
    const float* Wv   = (const float*)d_in[6];
    const float* bv   = (const float*)d_in[7];
    const float* Wo   = (const float*)d_in[8];
    const float* bo   = (const float*)d_in[9];
    const float* Wl   = (const float*)d_in[10];
    const float* bl   = (const float*)d_in[11];
    float* out = (float*)d_out;

    // ws layout (all 256B aligned): total ~8.06 MB
    char*   ws     = (char*)d_ws;
    __bf16* wkbf   = (__bf16*)(ws);                 // 768*768*2 = 1179648
    float*  qv     = (float*)(ws + 1179648);        // 768*4     = 3072
    float*  scores = (float*)(ws + 1182720);        // 16*12*8192*4 = 6291456
    float*  px     = (float*)(ws + 7474176);        // 16*12*768*4  = 589824

    k_cvt_wk <<<dim3((DIM * DIM + 255) / 256), dim3(256), 0, stream>>>(Wk, wkbf);
    k_q      <<<dim3(3),                       dim3(256), 0, stream>>>(pool, Wq, bq, qv);
    k_scores <<<dim3(SEQ / TS, NB),            dim3(256), 0, stream>>>(x, bk, wkbf, qv, scores);
    k_softmax<<<dim3(NB * NH),                 dim3(256), 0, stream>>>(scores);
    k_zero   <<<dim3((NB * NH * DIM + 255) / 256), dim3(256), 0, stream>>>(px, NB * NH * DIM);
    k_pool   <<<dim3(SEQ / 512, NB),           dim3(256), 0, stream>>>(x, scores, px);
    k_head   <<<dim3(NB),                      dim3(256), 0, stream>>>(px, Wv, bv, Wo, bo, Wl, bl, out);
}
